// MultiHeadAttention_81028853006766
// MI455X (gfx1250) — matrix operand order repaired and checked by reading, not compiled, not run on hardware
//
#include <hip/hip_runtime.h>
#include <hip/hip_bf16.h>
#include <stdint.h>

#define D_MODEL   1024
#define NUM_HEADS 16
#define D_K       64
#define BATCH     2
#define SEQ       2048

typedef __attribute__((ext_vector_type(16))) __bf16 v16bf;
typedef __attribute__((ext_vector_type(8)))  __bf16 v8bf;
typedef __attribute__((ext_vector_type(8)))  float  v8f;

static __device__ inline v8f wmma_bf16(v16bf a, v16bf b, v8f c) {
  // 8 args: (neg_a, A, neg_b, B, c_mod, C, reuse_a, reuse_b)
  return __builtin_amdgcn_wmma_f32_16x16x32_bf16(false, a, false, b, (short)0, c,
                                                 false, false);
}

// ---------------------------------------------------------------------------
// One-shot f32 -> bf16 conversion (memory bound; removes all cvt work from the
// GEMM inner loops). n must be a multiple of 8.
// ---------------------------------------------------------------------------
__global__ __launch_bounds__(256)
void cvt_f32_bf16_kernel(const float* __restrict__ src, __bf16* __restrict__ dst,
                         int n)
{
  const int i = (blockIdx.x * 256 + threadIdx.x) * 8;
  if (i >= n) return;
  const float4 a = *(const float4*)(src + i);
  const float4 b = *(const float4*)(src + i + 4);
  v8bf o;
  o[0] = (__bf16)a.x; o[1] = (__bf16)a.y; o[2] = (__bf16)a.z; o[3] = (__bf16)a.w;
  o[4] = (__bf16)b.x; o[5] = (__bf16)b.y; o[6] = (__bf16)b.z; o[7] = (__bf16)b.w;
  *(v8bf*)(dst + i) = o;
}

// ---------------------------------------------------------------------------
// GEMM:  Y[M,N] = A[M,K] @ W[N,K]^T (+ bias, * alpha), all-bf16 operands.
// A tile (64 rows x 32 K, 4KB) is staged into LDS with async global->LDS
// copies (double buffered, ASYNCcnt), shared by all 8 waves; B-fragments are
// direct 32B global loads. Wave tile 32(M) x 64(N) -> 8 WMMAs per K-step.
// Block: 256 thr = 8 waves (2 along M x 4 along N); block tile 64 x 256.
// mode 0: f32 row-major [M,N]
// mode 1: bf16 [B,H,T,dk]   (Q with alpha=1/sqrt(dk), K with alpha=1)
// mode 2: bf16 [B,H,dk,T]   (V transposed; packed b128 epilogue stores)
// ---------------------------------------------------------------------------
__global__ __launch_bounds__(256)
void gemm_bf16_kernel(const __bf16* __restrict__ A, const __bf16* __restrict__ W,
                      const float* __restrict__ bias, float alpha, int mode,
                      float* __restrict__ outF, __bf16* __restrict__ outB,
                      int M, int N, int K)
{
  __shared__ alignas(16) __bf16 Atile[2][64][32];   // 2 x 4KB double buffer

  const int tid  = threadIdx.x;
  const int lane = tid & 31;
  const int wave = tid >> 5;
  const int wm = wave & 1, wn = wave >> 1;
  const int blockM = blockIdx.x * 64;
  const int m0 = wm * 32;                       // wave M offset within block
  const int n0 = blockIdx.y * 256 + wn * 64;    // wave N offset (global)
  const int half = lane >> 4;
  const int colL = lane & 15;

  // Cooperative async-copy mapping: 256 lanes x 16B == 64 rows x 64B tile.
  const int crow   = tid >> 2;          // 0..63
  const int cchunk = (tid & 3) * 8;     // bf16 elems: 0,8,16,24
  const __bf16* gsrc = A + (size_t)(blockM + crow) * K + cchunk;

  auto stage = [&](int buf, int k0) {
    const uint32_t ldsa = (uint32_t)(uintptr_t)&Atile[buf][crow][cchunk];
    const uint64_t ga   = (uint64_t)(uintptr_t)(gsrc + k0);
    asm volatile("global_load_async_to_lds_b128 %0, %1, off"
                 :: "v"(ldsa), "v"(ga) : "memory");
  };

  v8f acc[2][4];
#pragma unroll
  for (int s = 0; s < 2; ++s)
#pragma unroll
    for (int j = 0; j < 4; ++j) acc[s][j] = (v8f){0,0,0,0,0,0,0,0};

  stage(0, 0);                                   // prologue fill

  for (int k0 = 0; k0 < K; k0 += 32) {
    const int buf = (k0 >> 5) & 1;
    if (k0 + 32 < K) {
      stage(buf ^ 1, k0 + 32);                   // prefetch next tile
      asm volatile("s_wait_asynccnt 1" ::: "memory");  // oldest (this buf) landed
    } else {
      asm volatile("s_wait_asynccnt 0" ::: "memory");
    }
    __syncthreads();                             // whole tile visible

    // A fragments from LDS (contiguous K order, matching B):
    //   lanes 0-15:  row=lane,    K +0..15
    //   lanes 16-31: row=lane-16, K +16..31
    v16bf af[2];
#pragma unroll
    for (int s = 0; s < 2; ++s) {
      const __bf16* p = &Atile[buf][m0 + s * 16 + colL][half << 4];
      v8bf lo = *(const v8bf*)p;
      v8bf hi = *(const v8bf*)(p + 8);
#pragma unroll
      for (int i = 0; i < 8; ++i) { af[s][i] = lo[i]; af[s][8 + i] = hi[i]; }
    }

    // B fragments direct from global (lane n holds col n; 16 contiguous K).
#pragma unroll
    for (int j = 0; j < 4; ++j) {
      const __bf16* q = W + (size_t)(n0 + j * 16 + colL) * K + k0 + (half << 4);
      if (k0 + 32 < K) __builtin_prefetch(q + 32, 0, 1);
      v16bf bfrag = *(const v16bf*)q;
      acc[0][j] = wmma_bf16(af[0], bfrag, acc[0][j]);
      acc[1][j] = wmma_bf16(af[1], bfrag, acc[1][j]);
    }

    __syncthreads();         // all waves done reading buf before it is refilled
  }

  // Epilogue. D-frag: lanes 0-15 VGPR r -> row r; lanes 16-31 -> row 8+r.
  const int rtile = 8 * half;
#pragma unroll
  for (int s = 0; s < 2; ++s) {
#pragma unroll
    for (int j = 0; j < 4; ++j) {
      const int col = n0 + j * 16 + colL;
      const float bv = bias[col];
      const int row0 = blockM + m0 + s * 16 + rtile;
      if (mode == 0) {
#pragma unroll
        for (int r = 0; r < 8; ++r)
          outF[(size_t)(row0 + r) * N + col] = (acc[s][j][r] + bv) * alpha;
      } else if (mode == 1) {
        const int h = col >> 6, d = col & 63;
        const int b = row0 / SEQ, t0 = row0 % SEQ;
        __bf16* o = outB + ((((size_t)b * NUM_HEADS + h) * SEQ + t0) * D_K + d);
#pragma unroll
        for (int r = 0; r < 8; ++r)
          o[(size_t)r * D_K] = (__bf16)((acc[s][j][r] + bv) * alpha);
      } else {
        const int h = col >> 6, d = col & 63;
        const int b = row0 / SEQ, t0 = row0 % SEQ;
        v8bf pk;
#pragma unroll
        for (int r = 0; r < 8; ++r) pk[r] = (__bf16)((acc[s][j][r] + bv) * alpha);
        *(v8bf*)(outB + ((((size_t)b * NUM_HEADS + h) * D_K + d) * SEQ + t0)) = pk;
      }
    }
  }
}

// ---------------------------------------------------------------------------
// Flash attention: one wave per 16-row Q tile; KV in chunks of 32.
// Q,K: bf16 [B,H,T,dk] (Q pre-scaled by 1/sqrt(dk)); V: bf16 [B,H,dk,T].
// O: bf16 [B,T,D] (feeds the bf16 output-projection GEMM directly).
// ---------------------------------------------------------------------------
__global__ __launch_bounds__(128)
void flash_attn_kernel(const __bf16* __restrict__ Q, const __bf16* __restrict__ Kt,
                       const __bf16* __restrict__ V, __bf16* __restrict__ O)
{
  __shared__ alignas(32) __bf16 plds[4][16][32];   // 1KB per wave (private)

  const int lane = threadIdx.x & 31;
  const int wave = threadIdx.x >> 5;
  const int half = lane >> 4;
  const int colL = lane & 15;

  const int bh = blockIdx.y;
  const int b  = bh >> 4;          // / NUM_HEADS
  const int h  = bh & 15;
  const int qrow0 = (blockIdx.x * 4 + wave) * 16;

  const __bf16* Qb = Q  + (size_t)bh * SEQ * D_K;
  const __bf16* Kb = Kt + (size_t)bh * SEQ * D_K;
  const __bf16* Vb = V  + (size_t)bh * D_K * SEQ;

  v16bf qa[2];
  {
    const __bf16* p = Qb + (size_t)(qrow0 + colL) * D_K + (half << 4);
#pragma unroll
    for (int kk = 0; kk < 2; ++kk) {
      v8bf lo = *(const v8bf*)(p + 32 * kk);
      v8bf hi = *(const v8bf*)(p + 32 * kk + 8);
#pragma unroll
      for (int i = 0; i < 8; ++i) { qa[kk][i] = lo[i]; qa[kk][8 + i] = hi[i]; }
    }
  }

  v8f oacc[4];
#pragma unroll
  for (int j = 0; j < 4; ++j) oacc[j] = (v8f){0,0,0,0,0,0,0,0};
  float mrow[8], lrow[8];
#pragma unroll
  for (int r = 0; r < 8; ++r) { mrow[r] = -__builtin_inff(); lrow[r] = 0.0f; }

  for (int kv0 = 0; kv0 < SEQ; kv0 += 32) {
    v8f s[2];
    s[0] = (v8f){0,0,0,0,0,0,0,0};
    s[1] = (v8f){0,0,0,0,0,0,0,0};
#pragma unroll
    for (int kk = 0; kk < 2; ++kk) {
#pragma unroll
      for (int j = 0; j < 2; ++j) {
        const __bf16* q = Kb + (size_t)(kv0 + j * 16 + colL) * D_K + kk * 32 + (half << 4);
        v16bf kf = *(const v16bf*)q;
        s[j] = wmma_bf16(qa[kk], kf, s[j]);
      }
    }

    // Online softmax; each row lives in a 16-lane half of the wave.
    float mcur[8];
#pragma unroll
    for (int r = 0; r < 8; ++r) mcur[r] = fmaxf(s[0][r], s[1][r]);
#pragma unroll
    for (int mk = 1; mk <= 8; mk <<= 1)
#pragma unroll
      for (int r = 0; r < 8; ++r)
        mcur[r] = fmaxf(mcur[r], __shfl_xor(mcur[r], mk, 32));

    float scale[8], rsum[8];
#pragma unroll
    for (int r = 0; r < 8; ++r) {
      const float mnew = fmaxf(mrow[r], mcur[r]);
      scale[r] = __expf(mrow[r] - mnew);
      mrow[r] = mnew;
      const float e0 = __expf(s[0][r] - mnew);
      const float e1 = __expf(s[1][r] - mnew);
      s[0][r] = e0; s[1][r] = e1;
      rsum[r] = e0 + e1;
    }
#pragma unroll
    for (int mk = 1; mk <= 8; mk <<= 1)
#pragma unroll
      for (int r = 0; r < 8; ++r)
        rsum[r] += __shfl_xor(rsum[r], mk, 32);
#pragma unroll
    for (int r = 0; r < 8; ++r) lrow[r] = lrow[r] * scale[r] + rsum[r];
#pragma unroll
    for (int j = 0; j < 4; ++j)
#pragma unroll
      for (int r = 0; r < 8; ++r) oacc[j][r] *= scale[r];

    // P (D-frag f32) -> bf16 A-frag via wave-private LDS (same-wave LDS ops
    // are in-order; no barrier needed).
#pragma unroll
    for (int r = 0; r < 8; ++r) {
      plds[wave][8 * half + r][colL]      = (__bf16)s[0][r];
      plds[wave][8 * half + r][16 + colL] = (__bf16)s[1][r];
    }
    v16bf pa;
    {
      const __bf16* p = &plds[wave][colL][half << 4];
      v8bf lo = *(const v8bf*)p;
      v8bf hi = *(const v8bf*)(p + 8);
#pragma unroll
      for (int i = 0; i < 8; ++i) { pa[i] = lo[i]; pa[8 + i] = hi[i]; }
    }

    // O[16x64] += P[16x32] @ V[32x64]  (V transposed -> contiguous B-frags)
#pragma unroll
    for (int j = 0; j < 4; ++j) {
      const __bf16* q = Vb + (size_t)(j * 16 + colL) * SEQ + kv0 + (half << 4);
      v16bf vf = *(const v16bf*)q;
      oacc[j] = wmma_bf16(pa, vf, oacc[j]);
    }
  }

  // Normalize + store O (bf16, [B,T,D], column block h*64).
#pragma unroll
  for (int r = 0; r < 8; ++r) {
    const int t = qrow0 + 8 * half + r;
    const float inv = 1.0f / lrow[r];
    __bf16* orow = O + ((size_t)b * SEQ + t) * D_MODEL + h * D_K;
#pragma unroll
    for (int j = 0; j < 4; ++j)
      orow[j * 16 + colL] = (__bf16)(oacc[j][r] * inv);
  }
}

// ---------------------------------------------------------------------------
extern "C" void kernel_launch(void* const* d_in, const int* in_sizes, int n_in,
                              void* d_out, int out_size, void* d_ws, size_t ws_size,
                              hipStream_t stream)
{
  const float* x  = (const float*)d_in[0];
  const float* wq = (const float*)d_in[1];
  const float* bq = (const float*)d_in[2];
  const float* wk = (const float*)d_in[3];
  const float* bk = (const float*)d_in[4];
  const float* wv = (const float*)d_in[5];
  const float* bv = (const float*)d_in[6];
  const float* wo = (const float*)d_in[7];
  const float* bo = (const float*)d_in[8];

  const int M = BATCH * SEQ;   // 4096
  const int N = D_MODEL;       // 1024
  const int K = D_MODEL;       // 1024
  const int nX = M * D_MODEL;          // 4M elems
  const int nW = D_MODEL * D_MODEL;    // 1M elems

  char* ws = (char*)d_ws;
  size_t off = 0;
  __bf16* xb  = (__bf16*)(ws + off); off += (size_t)nX * 2;                 // 8 MB
  __bf16* wqb = (__bf16*)(ws + off); off += (size_t)nW * 2;                 // 2 MB
  __bf16* wkb = (__bf16*)(ws + off); off += (size_t)nW * 2;
  __bf16* wvb = (__bf16*)(ws + off); off += (size_t)nW * 2;
  __bf16* wob = (__bf16*)(ws + off); off += (size_t)nW * 2;
  const size_t qkvElems = (size_t)BATCH * NUM_HEADS * SEQ * D_K;            // 4M
  __bf16* qbuf = (__bf16*)(ws + off); off += qkvElems * 2;                  // 8 MB
  __bf16* kbuf = (__bf16*)(ws + off); off += qkvElems * 2;
  __bf16* vbuf = (__bf16*)(ws + off); off += qkvElems * 2;
  __bf16* obuf = (__bf16*)(ws + off); off += (size_t)nX * 2;                // 8 MB

  // 1) One-shot bf16 conversions (memory bound).
  cvt_f32_bf16_kernel<<<nX / 8 / 256, 256, 0, stream>>>(x,  xb,  nX);
  cvt_f32_bf16_kernel<<<nW / 8 / 256, 256, 0, stream>>>(wq, wqb, nW);
  cvt_f32_bf16_kernel<<<nW / 8 / 256, 256, 0, stream>>>(wk, wkb, nW);
  cvt_f32_bf16_kernel<<<nW / 8 / 256, 256, 0, stream>>>(wv, wvb, nW);
  cvt_f32_bf16_kernel<<<nW / 8 / 256, 256, 0, stream>>>(wo, wob, nW);

  // 2) Q/K/V projections (Q absorbs 1/sqrt(dk) = 0.125).
  dim3 gblk(256), ggrid(M / 64, N / 256);
  gemm_bf16_kernel<<<ggrid, gblk, 0, stream>>>(xb, wqb, bq, 0.125f, 1, nullptr, qbuf, M, N, K);
  gemm_bf16_kernel<<<ggrid, gblk, 0, stream>>>(xb, wkb, bk, 1.0f,   1, nullptr, kbuf, M, N, K);
  gemm_bf16_kernel<<<ggrid, gblk, 0, stream>>>(xb, wvb, bv, 1.0f,   2, nullptr, vbuf, M, N, K);

  // 3) Flash attention.
  dim3 ablk(128), agrid(SEQ / 64, BATCH * NUM_HEADS);
  flash_attn_kernel<<<agrid, ablk, 0, stream>>>(qbuf, kbuf, vbuf, obuf);

  // 4) Output projection -> d_out (f32).
  gemm_bf16_kernel<<<ggrid, gblk, 0, stream>>>(obuf, wob, bo, 1.0f, 0, (float*)d_out, nullptr, M, N, K);
}
